// MoNet_13709535609127
// MI455X (gfx1250) — matrix operand order repaired and checked by reading, not compiled, not run on hardware
//
#include <hip/hip_runtime.h>

typedef _Float16 h16;
typedef __attribute__((ext_vector_type(16))) _Float16 v16h;
typedef __attribute__((ext_vector_type(8)))  float    v8f;

#define SCALE_F (2.0f*28.0f*0.32178f)
#define KGAUSS 9
#define CDIV(a,b) (((a)+(b)-1)/(b))

// ---------------- device helpers ----------------

// 16-bit A-matrix 16x32 K index for (lane, packed half j) per CDNA5 ISA 7.12.2
__device__ __forceinline__ int a_kidx(int lane, int j) {
  int v = j >> 1, b = j & 1;
  int hi = (lane >= 16) ? 8 : 0;
  return (v < 4) ? (hi + 2*v + b) : (16 + hi + 2*(v-4) + b);
}

__device__ __forceinline__ float eluf(float v) { return v > 0.f ? v : (__expf(v) - 1.f); }

// monotone float<->uint mapping for atomicMax on floats
__device__ __forceinline__ unsigned encf(float f) {
  unsigned u = __float_as_uint(f);
  return (u >> 31) ? ~u : (u | 0x80000000u);
}
__device__ __forceinline__ float decf(unsigned u) {
  return __uint_as_float((u >> 31) ? (u & 0x7FFFFFFFu) : ~u);
}

// w[k] = exp(-0.5 * (dx^2 * isg_x + dy^2 * isg_y)), isg = 1/sigma^2 precomputed
__device__ __forceinline__ void gmm_w(float ux, float uy,
                                      const float* __restrict__ mu,
                                      const float* __restrict__ isg,
                                      float* w) {
#pragma unroll
  for (int k = 0; k < KGAUSS; ++k) {
    float dx = ux - mu[2*k], dy = uy - mu[2*k+1];
    w[k] = __expf(-0.5f * (dx*dx*isg[2*k] + dy*dy*isg[2*k+1]));
  }
}

// tiny prep: isg[t] = 1 / sigma[t]^2  (t < 18)
__global__ void prep_isig(const float* __restrict__ sg, float* __restrict__ isg) {
  int t = threadIdx.x;
  if (t < 2 * KGAUSS) {
    float s = sg[t];
    isg[t] = 1.0f / (s * s);
  }
}

// ---------------- layer 1 (IN=1, scalar) ----------------

__global__ __launch_bounds__(256)
void layer1_edge(const int* __restrict__ ei, int E,
                 const float* __restrict__ posv, const float* __restrict__ x,
                 const float* __restrict__ g1, const float* __restrict__ mu,
                 const float* __restrict__ isg,
                 float* __restrict__ agg, float* __restrict__ cnt) {
  int e = blockIdx.x * blockDim.x + threadIdx.x;
  if (e >= E) return;
  int r = ei[e], c = ei[E + e];
  float ux = (posv[2*c]   - posv[2*r])   * (1.f/SCALE_F) + 0.5f;
  float uy = (posv[2*c+1] - posv[2*r+1]) * (1.f/SCALE_F) + 0.5f;
  float w[KGAUSS];
  gmm_w(ux, uy, mu, isg, w);
  float xs = x[r];
  atomicAdd(&cnt[c], 1.0f);
  for (int o = 0; o < 32; ++o) {
    float s = 0.f;
#pragma unroll
    for (int k = 0; k < KGAUSS; ++k) s += g1[k*32 + o] * w[k];
    atomicAdd(&agg[c*32 + o], xs * s);
  }
}

__global__ __launch_bounds__(256)
void finish1(const float* __restrict__ agg, const float* __restrict__ cnt,
             const float* __restrict__ x, const float* __restrict__ root,
             const float* __restrict__ b, float* __restrict__ h, int N) {
  int t = blockIdx.x * blockDim.x + threadIdx.x;
  if (t >= N * 32) return;
  int n = t >> 5, o = t & 31;
  float v = agg[t] / fmaxf(cnt[n], 1.f) + x[n] * root[o] + b[o];
  h[t] = eluf(v);
}

// ---------------- pooling (segment max + pos mean) ----------------

template<int C>
__global__ __launch_bounds__(256)
void pool_max(const float* __restrict__ h, const int* __restrict__ cl,
              const float* __restrict__ posin,
              unsigned* __restrict__ hmax, float* __restrict__ psum,
              float* __restrict__ pcnt, int N) {
  int n = blockIdx.x * blockDim.x + threadIdx.x;
  if (n >= N) return;
  int c = cl[n];
  for (int o = 0; o < C; ++o) atomicMax(&hmax[c*C + o], encf(h[n*C + o]));
  atomicAdd(&psum[2*c],     posin[2*n]);
  atomicAdd(&psum[2*c + 1], posin[2*n + 1]);
  atomicAdd(&pcnt[c], 1.0f);
}

template<int C>
__global__ __launch_bounds__(256)
void pool_finish(const unsigned* __restrict__ hmax, const float* __restrict__ psum,
                 const float* __restrict__ pcnt, float* __restrict__ hout,
                 float* __restrict__ posout, int M) {
  int t = blockIdx.x * blockDim.x + threadIdx.x;
  if (t >= M * C) return;
  int n = t / C, o = t % C;
  hout[t] = decf(hmax[t]);
  if (o < 2) posout[2*n + o] = psum[2*n + o] / fmaxf(pcnt[n], 1.0f);
}

// ---------------- B-matrix fragment prep: g(i,k,o) f32 -> f16 frag order ----

template<int IN>
__global__ __launch_bounds__(256)
void prep_g(const float* __restrict__ g, h16* __restrict__ gH) {
  constexpr int KT = IN * KGAUSS / 32;
  int idx = blockIdx.x * blockDim.x + threadIdx.x;
  if (idx >= KT * 4 * 32 * 16) return;
  int j    = idx & 15;
  int lane = (idx >> 4) & 31;
  int nt   = (idx >> 9) & 3;
  int kt   = idx >> 11;
  int k    = (IN == 32) ? kt : (kt >> 1);
  int ioff = (IN == 32) ? 0 : ((kt & 1) * 32);
  int i    = ioff + ((j < 8) ? ((lane >> 4) * 8 + j) : (16 + (lane >> 4) * 8 + (j - 8)));   // B-fragment K row
  int o    = nt * 16 + (lane & 15);           // B-fragment N col
  gH[idx] = (h16)g[(i * KGAUSS + k) * 64 + o];
}

// ---------------- WMMA edge kernel: msg = U @ G, scatter-add by col --------

template<int IN, bool USE_LDS>
__global__ __launch_bounds__(128)
void gmm_edge_wmma(const int* __restrict__ ei, int E,
                   const float* __restrict__ posv,
                   const float* __restrict__ hin,
                   const h16* __restrict__ gH,
                   const float* __restrict__ mu,
                   const float* __restrict__ isg,
                   float* __restrict__ agg,
                   float* __restrict__ cnt) {
  constexpr int KT = IN * KGAUSS / 32;      // 9 (IN=32) or 18 (IN=64)
  constexpr int GH = KT * 4 * 32 * 16;      // halves
  __shared__ h16 lds[USE_LDS ? GH : 16];
  if constexpr (USE_LDS) {
    const uint4* s = (const uint4*)gH;
    uint4* d = (uint4*)lds;
    for (int i = threadIdx.x; i < GH / 8; i += blockDim.x) d[i] = s[i];
    __syncthreads();
  }

  const int lane = threadIdx.x & 31;
  const int wave = threadIdx.x >> 5;
  const int base = blockIdx.x * 64 + wave * 16;   // 16 edges per wave
  const int em   = base + (lane & 15);
  const bool valid = em < E;
  const int e = valid ? em : (E - 1);
  const int r = ei[e];
  const int c = ei[E + e];

  float ux = (posv[2*c]   - posv[2*r])   * (1.f/SCALE_F) + 0.5f;
  float uy = (posv[2*c+1] - posv[2*r+1]) * (1.f/SCALE_F) + 0.5f;
  float w[KGAUSS];
  gmm_w(ux, uy, mu, isg, w);
#pragma unroll
  for (int k = 0; k < KGAUSS; ++k) if (!valid) w[k] = 0.f;

  // per-lane h values (16 per 32-wide i-block), loaded once, reused all K-tiles
  v16h hv[IN / 32];
#pragma unroll
  for (int blk = 0; blk < IN / 32; ++blk) {
#pragma unroll
    for (int j = 0; j < 16; ++j)
      hv[blk][j] = (h16)hin[r * IN + blk * 32 + a_kidx(lane, j)];
  }

  v8f z = {0.f,0.f,0.f,0.f,0.f,0.f,0.f,0.f};
  v8f acc[4] = {z, z, z, z};

#pragma unroll
  for (int kt = 0; kt < KT; ++kt) {
    const int k = (IN == 32) ? kt : (kt >> 1);
    const h16 wh = (h16)w[k];
    v16h a;
#pragma unroll
    for (int j = 0; j < 16; ++j) a[j] = hv[(IN == 32) ? 0 : (kt & 1)][j] * wh;
#pragma unroll
    for (int nt = 0; nt < 4; ++nt) {
      const int boff = ((kt * 4 + nt) * 32 + lane) * 16;
      v16h b;
      if constexpr (USE_LDS) b = *(const v16h*)&lds[boff];
      else                   b = *(const v16h*)&gH[boff];
      acc[nt] = __builtin_amdgcn_wmma_f32_16x16x32_f16(
          false, a, false, b, (short)0, acc[nt], false, false);
    }
  }

  // C/D layout: lane holds N = lane%16; VGPR r -> M = r + (lane>=16 ? 8 : 0)
  const int mrow = (lane >= 16) ? 8 : 0;
  int cols[8];
#pragma unroll
  for (int rr = 0; rr < 8; ++rr) {
    int ee = base + mrow + rr;
    cols[rr] = (ee < E) ? ei[E + ee] : -1;
  }
#pragma unroll
  for (int nt = 0; nt < 4; ++nt) {
    const int o = nt * 16 + (lane & 15);
#pragma unroll
    for (int rr = 0; rr < 8; ++rr) {
      if (cols[rr] >= 0) atomicAdd(&agg[cols[rr] * 64 + o], acc[nt][rr]);
    }
  }
  if ((lane < 16) && valid) atomicAdd(&cnt[c], 1.0f);
}

// ---------------- finish: agg/cnt + h@root + b, ELU ----------------

template<int IN>
__global__ __launch_bounds__(256)
void finish_root(const float* __restrict__ agg, const float* __restrict__ cnt,
                 const float* __restrict__ hin, const float* __restrict__ root,
                 const float* __restrict__ b, float* __restrict__ hout, int M) {
  int t = blockIdx.x * blockDim.x + threadIdx.x;
  if (t >= M * 64) return;
  int n = t >> 6, o = t & 63;
  float s = agg[t] / fmaxf(cnt[n], 1.0f) + b[o];
#pragma unroll 8
  for (int i = 0; i < IN; ++i) s += hin[n * IN + i] * root[i * 64 + o];
  hout[t] = eluf(s);
}

// ---------------- global mean + FC head ----------------

__global__ __launch_bounds__(256)
void gsum_kernel(const float* __restrict__ h, float* __restrict__ acc, int M) {
  int t = blockIdx.x * blockDim.x + threadIdx.x;
  if (t >= M * 64) return;
  atomicAdd(&acc[t & 63], h[t]);
}

__global__ __launch_bounds__(128)
void final_fc(const float* __restrict__ gs, float invM,
              const float* __restrict__ fw, const float* __restrict__ fb,
              float* __restrict__ out) {
  int j = threadIdx.x;
  if (j >= 128) return;
  float s = fb[j];
#pragma unroll 8
  for (int i = 0; i < 64; ++i) s += gs[i] * invM * fw[i * 128 + j];
  out[j] = s;
}

// ---------------- host ----------------

extern "C" void kernel_launch(void* const* d_in, const int* in_sizes, int n_in,
                              void* d_out, int out_size, void* d_ws, size_t ws_size,
                              hipStream_t stream) {
  const float* x      = (const float*)d_in[0];
  const float* pos    = (const float*)d_in[1];
  const int*   ei1    = (const int*)  d_in[2];
  const int*   cl1    = (const int*)  d_in[3];
  const int*   ei2    = (const int*)  d_in[4];
  const int*   cl2    = (const int*)  d_in[5];
  const int*   ei3    = (const int*)  d_in[6];
  const float* g1     = (const float*)d_in[8];
  const float* mu1    = (const float*)d_in[9];
  const float* sig1   = (const float*)d_in[10];
  const float* root1  = (const float*)d_in[11];
  const float* b1     = (const float*)d_in[12];
  const float* g2     = (const float*)d_in[13];
  const float* mu2    = (const float*)d_in[14];
  const float* sig2   = (const float*)d_in[15];
  const float* root2  = (const float*)d_in[16];
  const float* b2     = (const float*)d_in[17];
  const float* g3     = (const float*)d_in[18];
  const float* mu3    = (const float*)d_in[19];
  const float* sig3   = (const float*)d_in[20];
  const float* root3  = (const float*)d_in[21];
  const float* b3     = (const float*)d_in[22];
  const float* fw     = (const float*)d_in[23];
  const float* fb     = (const float*)d_in[24];
  float* out = (float*)d_out;

  const int N  = in_sizes[0];
  const int E1 = in_sizes[2] / 2;
  const int C1 = in_sizes[5];
  const int E2 = in_sizes[4] / 2;
  const int C2 = in_sizes[7];
  const int E3 = in_sizes[6] / 2;

  char* wsb = (char*)d_ws;
  size_t off = 0;
  auto alloc = [&](size_t bytes) -> void* {
    void* p = wsb + off;
    off = (off + bytes + 255) & ~(size_t)255;
    return p;
  };

  float*    agg1  = (float*)   alloc((size_t)N * 32 * 4);
  float*    cnt1  = (float*)   alloc((size_t)N * 4);
  float*    h1    = (float*)   alloc((size_t)N * 32 * 4);
  unsigned* hp1u  = (unsigned*)alloc((size_t)C1 * 32 * 4);
  float*    psum1 = (float*)   alloc((size_t)C1 * 2 * 4);
  float*    pcnt1 = (float*)   alloc((size_t)C1 * 4);
  float*    hp1f  = (float*)   alloc((size_t)C1 * 32 * 4);
  float*    pos2  = (float*)   alloc((size_t)C1 * 2 * 4);
  h16*      g2h   = (h16*)     alloc((size_t)9 * 4 * 32 * 16 * 2);
  float*    agg2  = (float*)   alloc((size_t)C1 * 64 * 4);
  float*    cnt2  = (float*)   alloc((size_t)C1 * 4);
  float*    h2    = (float*)   alloc((size_t)C1 * 64 * 4);
  unsigned* hp2u  = (unsigned*)alloc((size_t)C2 * 64 * 4);
  float*    psum2 = (float*)   alloc((size_t)C2 * 2 * 4);
  float*    pcnt2 = (float*)   alloc((size_t)C2 * 4);
  float*    hp2f  = (float*)   alloc((size_t)C2 * 64 * 4);
  float*    pos3  = (float*)   alloc((size_t)C2 * 2 * 4);
  h16*      g3h   = (h16*)     alloc((size_t)18 * 4 * 32 * 16 * 2);
  float*    agg3  = (float*)   alloc((size_t)C2 * 64 * 4);
  float*    cnt3  = (float*)   alloc((size_t)C2 * 4);
  float*    h3    = (float*)   alloc((size_t)C2 * 64 * 4);
  float*    gs    = (float*)   alloc((size_t)64 * 4);
  float*    isig1 = (float*)   alloc((size_t)2 * KGAUSS * 4);
  float*    isig2 = (float*)   alloc((size_t)2 * KGAUSS * 4);
  float*    isig3 = (float*)   alloc((size_t)2 * KGAUSS * 4);

  // ---- parameter prep (1/sigma^2) ----
  prep_isig<<<1, 32, 0, stream>>>(sig1, isig1);
  prep_isig<<<1, 32, 0, stream>>>(sig2, isig2);
  prep_isig<<<1, 32, 0, stream>>>(sig3, isig3);

  // ---- layer 1 ----
  hipMemsetAsync(agg1, 0, (size_t)N * 32 * 4, stream);
  hipMemsetAsync(cnt1, 0, (size_t)N * 4, stream);
  layer1_edge<<<CDIV(E1, 256), 256, 0, stream>>>(ei1, E1, pos, x, g1, mu1, isig1, agg1, cnt1);
  finish1<<<CDIV(N * 32, 256), 256, 0, stream>>>(agg1, cnt1, x, root1, b1, h1, N);

  // ---- pool 1 ----
  hipMemsetAsync(hp1u, 0, (size_t)C1 * 32 * 4, stream);
  hipMemsetAsync(psum1, 0, (size_t)C1 * 2 * 4, stream);
  hipMemsetAsync(pcnt1, 0, (size_t)C1 * 4, stream);
  pool_max<32><<<CDIV(N, 256), 256, 0, stream>>>(h1, cl1, pos, hp1u, psum1, pcnt1, N);
  pool_finish<32><<<CDIV(C1 * 32, 256), 256, 0, stream>>>(hp1u, psum1, pcnt1, hp1f, pos2, C1);

  // ---- layer 2 (WMMA, B in LDS) ----
  prep_g<32><<<CDIV(9 * 4 * 32 * 16, 256), 256, 0, stream>>>(g2, g2h);
  hipMemsetAsync(agg2, 0, (size_t)C1 * 64 * 4, stream);
  hipMemsetAsync(cnt2, 0, (size_t)C1 * 4, stream);
  gmm_edge_wmma<32, true><<<CDIV(E2, 64), 128, 0, stream>>>(
      ei2, E2, pos2, hp1f, g2h, mu2, isig2, agg2, cnt2);
  finish_root<32><<<CDIV(C1 * 64, 256), 256, 0, stream>>>(agg2, cnt2, hp1f, root2, b2, h2, C1);

  // ---- pool 2 ----
  hipMemsetAsync(hp2u, 0, (size_t)C2 * 64 * 4, stream);
  hipMemsetAsync(psum2, 0, (size_t)C2 * 2 * 4, stream);
  hipMemsetAsync(pcnt2, 0, (size_t)C2 * 4, stream);
  pool_max<64><<<CDIV(C1, 256), 256, 0, stream>>>(h2, cl2, pos2, hp2u, psum2, pcnt2, C1);
  pool_finish<64><<<CDIV(C2 * 64, 256), 256, 0, stream>>>(hp2u, psum2, pcnt2, hp2f, pos3, C2);

  // ---- layer 3 (WMMA, B streamed from L2) ----
  prep_g<64><<<CDIV(18 * 4 * 32 * 16, 256), 256, 0, stream>>>(g3, g3h);
  hipMemsetAsync(agg3, 0, (size_t)C2 * 64 * 4, stream);
  hipMemsetAsync(cnt3, 0, (size_t)C2 * 4, stream);
  gmm_edge_wmma<64, false><<<CDIV(E3, 64), 128, 0, stream>>>(
      ei3, E3, pos3, hp2f, g3h, mu3, isig3, agg3, cnt3);
  finish_root<64><<<CDIV(C2 * 64, 256), 256, 0, stream>>>(agg3, cnt3, hp2f, root3, b3, h3, C2);

  // ---- global mean + FC ----
  hipMemsetAsync(gs, 0, 64 * 4, stream);
  gsum_kernel<<<CDIV(C2 * 64, 256), 256, 0, stream>>>(h3, gs, C2);
  final_fc<<<1, 128, 0, stream>>>(gs, 1.0f / (float)C2, fw, fb, out);
}
